// PicoTrainableBlock_35631048687653
// MI455X (gfx1250) — matrix operand order repaired and checked by reading, not compiled, not run on hardware
//
#include <hip/hip_runtime.h>
#include <hip/hip_bf16.h>
#include <math.h>

typedef __bf16 bf16_t;
typedef __attribute__((ext_vector_type(16))) __bf16 v16bf;
typedef __attribute__((ext_vector_type(8)))  float  v8f;

union Frag { uint4 u[2]; v16bf v; };

#define Bdim 2
#define Sdim 2048
#define Ddim 1024
#define Hdim 16
#define HDdim 64
#define MLPdim 4096

static __device__ __forceinline__ v8f wmma_bf16(v16bf a, v16bf b, v8f c) {
  return __builtin_amdgcn_wmma_f32_16x16x32_bf16(false, a, false, b, (short)0, c, false, false);
}

// ---------------------------------------------------------------- cast f32->bf16
__global__ __launch_bounds__(256) void cast_bf16_k(const float* __restrict__ in,
                                                   bf16_t* __restrict__ out, int n) {
  int i = (blockIdx.x * 256 + threadIdx.x) * 4;
  if (i + 3 < n) {
    float4 f = *(const float4*)(in + i);
    out[i + 0] = (bf16_t)f.x; out[i + 1] = (bf16_t)f.y;
    out[i + 2] = (bf16_t)f.z; out[i + 3] = (bf16_t)f.w;
  }
}

// ---------------------------------------------------------------- layernorm -> bf16
__global__ __launch_bounds__(256) void ln_bf16_k(const float* __restrict__ x,
                                                 const float* __restrict__ g,
                                                 const float* __restrict__ bta,
                                                 bf16_t* __restrict__ out) {
  __shared__ float s1[256], s2[256];
  int row = blockIdx.x, tid = threadIdx.x;
  const float* xr = x + (size_t)row * Ddim;
  float v[4], s = 0.f, q = 0.f;
#pragma unroll
  for (int j = 0; j < 4; ++j) { v[j] = xr[tid + j * 256]; s += v[j]; q += v[j] * v[j]; }
  s1[tid] = s; s2[tid] = q;
  __syncthreads();
  for (int off = 128; off > 0; off >>= 1) {
    if (tid < off) { s1[tid] += s1[tid + off]; s2[tid] += s2[tid + off]; }
    __syncthreads();
  }
  float mu = s1[0] * (1.f / Ddim);
  float var = s2[0] * (1.f / Ddim) - mu * mu;
  float rs = rsqrtf(var + 1e-5f);
#pragma unroll
  for (int j = 0; j < 4; ++j) {
    int c = tid + j * 256;
    out[(size_t)row * Ddim + c] = (bf16_t)((v[j] - mu) * rs * g[c] + bta[c]);
  }
}

// ---------------------------------------------------------------- generic bf16 WMMA GEMM
// C[M,N] = A[M,K] @ B[K,N] (+bias)(+resid)(gelu), out fp32/bf16 -- all compile-time
#define BM 128
#define BN 128
#define BKt 32
#define LDT 40   // 32 + 8 pad (bf16 elems)

static __device__ __forceinline__ float gelu_tanh(float v) {
  float u = 0.7978845608028654f * (v + 0.044715f * v * v * v);
  return 0.5f * v * (1.f + tanhf(u));
}

template <int HAS_BIAS, int HAS_RES, int DO_GELU, int OUT_F, int OUT_B>
__global__ __launch_bounds__(256) void gemm_bf16_k(
    const bf16_t* __restrict__ A, const bf16_t* __restrict__ Bm,
    int M, int N, int K,
    const float* __restrict__ bias, const float* __restrict__ resid,
    float* __restrict__ outF, bf16_t* __restrict__ outB) {
  __shared__ bf16_t As[BM * LDT];
  __shared__ bf16_t Bs[BN * LDT];
  int tid = threadIdx.x;
  int lane = tid & 31, wid = tid >> 5;
  int half = lane >> 4, l16 = lane & 15;
  int wm = wid & 3, wn = wid >> 2;   // 4 (M) x 2 (N) waves
  int m0 = blockIdx.y * BM, n0 = blockIdx.x * BN;

  // per-thread global-load coordinates (2 x 16B chunks each for A and B)
  int arow0 = tid >> 2,           acol0 = (tid & 3) * 8;
  int arow1 = (tid + 256) >> 2,   acol1 = ((tid + 256) & 3) * 8;
  int bkr0 = tid >> 4,            bnc0 = (tid & 15) * 8;
  int bkr1 = (tid + 256) >> 4,    bnc1 = ((tid + 256) & 15) * 8;
  const bf16_t* aP0 = A + (size_t)(m0 + arow0) * K + acol0;
  const bf16_t* aP1 = A + (size_t)(m0 + arow1) * K + acol1;
  const bf16_t* bP0 = Bm + (size_t)bkr0 * N + n0 + bnc0;
  const bf16_t* bP1 = Bm + (size_t)bkr1 * N + n0 + bnc1;

  v8f acc[2][4] = {};

  // prologue: load chunk 0
  uint4 au0 = *(const uint4*)(aP0);
  uint4 au1 = *(const uint4*)(aP1);
  uint4 bu0 = *(const uint4*)(bP0);
  uint4 bu1 = *(const uint4*)(bP1);
  *(uint4*)(&As[arow0 * LDT + acol0]) = au0;
  *(uint4*)(&As[arow1 * LDT + acol1]) = au1;
  {
    const bf16_t* p0 = (const bf16_t*)&bu0;
    const bf16_t* p1 = (const bf16_t*)&bu1;
#pragma unroll
    for (int j = 0; j < 8; ++j) Bs[(bnc0 + j) * LDT + bkr0] = p0[j];
#pragma unroll
    for (int j = 0; j < 8; ++j) Bs[(bnc1 + j) * LDT + bkr1] = p1[j];
  }
  __syncthreads();

  int nk = K / BKt;
  for (int ki = 0; ki < nk; ++ki) {
    // prefetch next chunk's global data into registers (overlaps with WMMA)
    if (ki + 1 < nk) {
      int ko = (ki + 1) * BKt;
      au0 = *(const uint4*)(aP0 + ko);
      au1 = *(const uint4*)(aP1 + ko);
      bu0 = *(const uint4*)(bP0 + (size_t)ko * N);
      bu1 = *(const uint4*)(bP1 + (size_t)ko * N);
      if (ki + 2 < nk) {
        int kp = (ki + 2) * BKt;
        __builtin_prefetch(aP0 + kp, 0, 1);
        __builtin_prefetch(bP0 + (size_t)kp * N, 0, 1);
      }
    }

    Frag af[2];
#pragma unroll
    for (int fi = 0; fi < 2; ++fi) {
      const bf16_t* pa = &As[(wm * 32 + fi * 16 + l16) * LDT + half * 8];
      af[fi].u[0] = *(const uint4*)pa;
      af[fi].u[1] = *(const uint4*)(pa + 16);
    }
    Frag bfr[4];
#pragma unroll
    for (int fj = 0; fj < 4; ++fj) {
      const bf16_t* pb = &Bs[(wn * 64 + fj * 16 + l16) * LDT + half * 8];
      bfr[fj].u[0] = *(const uint4*)pb;
      bfr[fj].u[1] = *(const uint4*)(pb + 16);
    }
#pragma unroll
    for (int fi = 0; fi < 2; ++fi)
#pragma unroll
      for (int fj = 0; fj < 4; ++fj)
        acc[fi][fj] = wmma_bf16(af[fi].v, bfr[fj].v, acc[fi][fj]);

    __syncthreads();
    if (ki + 1 < nk) {
      *(uint4*)(&As[arow0 * LDT + acol0]) = au0;
      *(uint4*)(&As[arow1 * LDT + acol1]) = au1;
      const bf16_t* p0 = (const bf16_t*)&bu0;
      const bf16_t* p1 = (const bf16_t*)&bu1;
#pragma unroll
      for (int j = 0; j < 8; ++j) Bs[(bnc0 + j) * LDT + bkr0] = p0[j];
#pragma unroll
      for (int j = 0; j < 8; ++j) Bs[(bnc1 + j) * LDT + bkr1] = p1[j];
      __syncthreads();
    }
  }

#pragma unroll
  for (int fi = 0; fi < 2; ++fi)
#pragma unroll
    for (int fj = 0; fj < 4; ++fj) {
      int gn = n0 + wn * 64 + fj * 16 + l16;
      float bv = HAS_BIAS ? bias[gn] : 0.f;
#pragma unroll
      for (int r = 0; r < 8; ++r) {
        int gm = m0 + wm * 32 + fi * 16 + r + 8 * half;
        float v = acc[fi][fj][r] + bv;
        if (HAS_RES) v += resid[(size_t)gm * N + gn];
        if (DO_GELU) v = gelu_tanh(v);
        if (OUT_F) outF[(size_t)gm * N + gn] = v;
        if (OUT_B) outB[(size_t)gm * N + gn] = (bf16_t)v;
      }
    }
}

// ---------------------------------------------------------------- attention
// One workgroup: one (b,h), 16 query rows. Full 16x2048 score row lives in LDS.
#define SROW (Sdim + 8)

__global__ __launch_bounds__(256) void attn_k(const bf16_t* __restrict__ Q,
                                              const bf16_t* __restrict__ Kg,
                                              const bf16_t* __restrict__ V,
                                              float* __restrict__ probs,
                                              bf16_t* __restrict__ attn_out) {
  extern __shared__ char smem_raw[];
  float*  sc      = (float*)smem_raw;                                     // 16*SROW f32
  bf16_t* vts_all = (bf16_t*)(smem_raw + 16 * SROW * 4);                  // 8 * 64*40 bf16
  float*  red     = (float*)(smem_raw + 16 * SROW * 4 + 8 * 64 * 40 * 2); // 16*64 f32

  int tid = threadIdx.x, lane = tid & 31, wid = tid >> 5;
  int half = lane >> 4, l16 = lane & 15;
  int qt = blockIdx.x;               // 0..127
  int bh = blockIdx.y;               // 0..31
  int b = bh >> 4, h = bh & 15;
  const float NEG = -__builtin_inff();

  // Q fragments (held for whole kernel): rows qt*16+l16, hd contiguous
  const bf16_t* qrow = Q + (((size_t)(b * Sdim + qt * 16 + l16)) * Ddim + h * HDdim);
  Frag a0, a1;
  a0.u[0] = *(const uint4*)(qrow + half * 8);
  a0.u[1] = *(const uint4*)(qrow + 16 + half * 8);
  a1.u[0] = *(const uint4*)(qrow + 32 + half * 8);
  a1.u[1] = *(const uint4*)(qrow + 48 + half * 8);

  // ---- phase 1: scores = scale * Q K^T with causal mask -> LDS
  for (int kb = wid; kb < Sdim / 16; kb += 8) {
    int col = kb * 16 + l16;
    if (kb > qt) {
#pragma unroll
      for (int r = 0; r < 8; ++r) sc[(r + 8 * half) * SROW + col] = NEG;
    } else {
      const bf16_t* krow = Kg + (((size_t)(b * Sdim + kb * 16 + l16)) * Ddim + h * HDdim);
      Frag b0, b1;
      b0.u[0] = *(const uint4*)(krow + half * 8);
      b0.u[1] = *(const uint4*)(krow + 16 + half * 8);
      b1.u[0] = *(const uint4*)(krow + 32 + half * 8);
      b1.u[1] = *(const uint4*)(krow + 48 + half * 8);
      v8f c = {};
      c = wmma_bf16(a0.v, b0.v, c);
      c = wmma_bf16(a1.v, b1.v, c);
#pragma unroll
      for (int r = 0; r < 8; ++r) {
        int m = r + 8 * half;
        int qi = qt * 16 + m;
        sc[m * SROW + col] = (col <= qi) ? c[r] * 0.125f : NEG;
      }
    }
  }
  __syncthreads();

  // ---- phase 2: row softmax (wave handles 2 rows, wave32 shfl reductions)
  for (int rr = 0; rr < 2; ++rr) {
    int r = wid * 2 + rr;
    float* row = sc + r * SROW;
    float m = NEG;
    for (int k = lane; k < Sdim; k += 32) m = fmaxf(m, row[k]);
#pragma unroll
    for (int off = 16; off > 0; off >>= 1) m = fmaxf(m, __shfl_xor(m, off, 32));
    float sum = 0.f;
    for (int k = lane; k < Sdim; k += 32) {
      float e = __expf(row[k] - m);
      row[k] = e; sum += e;
    }
#pragma unroll
    for (int off = 16; off > 0; off >>= 1) sum += __shfl_xor(sum, off, 32);
    float inv = 1.f / sum;
    size_t ob = ((size_t)bh * Sdim + (qt * 16 + r)) * Sdim;
    for (int k = lane; k < Sdim; k += 32) {
      float p = row[k] * inv;
      row[k] = p;
      probs[ob + k] = p;
    }
  }
  __syncthreads();

  // ---- phase 3: attn_out = P @ V  (each wave owns 256 keys, reduce across waves)
  v8f oacc[4] = {};
  bf16_t* vts = vts_all + wid * (64 * 40);
  for (int c = 0; c < 8; ++c) {
    int kbase = wid * 256 + c * 32;
    // stage V chunk transposed: vts[hd][kk], lane = key kk
    const bf16_t* vrow = V + (((size_t)(b * Sdim + kbase + lane)) * Ddim + h * HDdim);
#pragma unroll
    for (int j8 = 0; j8 < 8; ++j8) {
      uint4 u = *(const uint4*)(vrow + j8 * 8);
      const bf16_t* pv = (const bf16_t*)&u;
#pragma unroll
      for (int j = 0; j < 8; ++j) vts[(j8 * 8 + j) * 40 + lane] = pv[j];
    }
    __syncthreads();
    // A fragment from probs in LDS (convert f32 -> bf16 in-register)
    const float* prow = sc + l16 * SROW + kbase;
    float4 f0 = *(const float4*)(prow + half * 8);
    float4 f1 = *(const float4*)(prow + half * 8 + 4);
    float4 g0 = *(const float4*)(prow + 16 + half * 8);
    float4 g1 = *(const float4*)(prow + 16 + half * 8 + 4);
    v16bf pa;
    pa[0] = (bf16_t)f0.x;  pa[1] = (bf16_t)f0.y;  pa[2]  = (bf16_t)f0.z;  pa[3]  = (bf16_t)f0.w;
    pa[4] = (bf16_t)f1.x;  pa[5] = (bf16_t)f1.y;  pa[6]  = (bf16_t)f1.z;  pa[7]  = (bf16_t)f1.w;
    pa[8] = (bf16_t)g0.x;  pa[9] = (bf16_t)g0.y;  pa[10] = (bf16_t)g0.z;  pa[11] = (bf16_t)g0.w;
    pa[12] = (bf16_t)g1.x; pa[13] = (bf16_t)g1.y; pa[14] = (bf16_t)g1.z;  pa[15] = (bf16_t)g1.w;
#pragma unroll
    for (int fj = 0; fj < 4; ++fj) {
      const bf16_t* pb = &vts[(fj * 16 + l16) * 40 + half * 8];
      Frag bfr;
      bfr.u[0] = *(const uint4*)pb;
      bfr.u[1] = *(const uint4*)(pb + 16);
      oacc[fj] = wmma_bf16(pa, bfr.v, oacc[fj]);
    }
    __syncthreads();
  }

  // cross-wave reduction in LDS
  for (int i = tid; i < 16 * 64; i += 256) red[i] = 0.f;
  __syncthreads();
#pragma unroll
  for (int fj = 0; fj < 4; ++fj)
#pragma unroll
    for (int r = 0; r < 8; ++r)
      atomicAdd(&red[(r + 8 * half) * 64 + fj * 16 + l16], oacc[fj][r]);
  __syncthreads();
  for (int i = tid; i < 16 * 64; i += 256) {
    int m = i >> 6, hd = i & 63;
    attn_out[((size_t)(b * Sdim + qt * 16 + m)) * Ddim + h * HDdim + hd] = (bf16_t)red[i];
  }
}

// ---------------------------------------------------------------- launch
extern "C" void kernel_launch(void* const* d_in, const int* in_sizes, int n_in,
                              void* d_out, int out_size, void* d_ws, size_t ws_size,
                              hipStream_t stream) {
  const float* x     = (const float*)d_in[0];
  const float* ln1_g = (const float*)d_in[1];
  const float* ln1_b = (const float*)d_in[2];
  const float* Wq    = (const float*)d_in[3];
  const float* Wk    = (const float*)d_in[4];
  const float* Wv    = (const float*)d_in[5];
  const float* Wo    = (const float*)d_in[6];
  const float* ln2_g = (const float*)d_in[7];
  const float* ln2_b = (const float*)d_in[8];
  const float* fc1_w = (const float*)d_in[9];
  const float* fc1_b = (const float*)d_in[10];
  const float* fc2_w = (const float*)d_in[11];
  const float* fc2_b = (const float*)d_in[12];

  const size_t T = (size_t)Bdim * Sdim;        // 4096 tokens
  char* w = (char*)d_ws;
  bf16_t* h1    = (bf16_t*)(w);                         // T*D bf16
  bf16_t* Wqb   = (bf16_t*)(w + 8388608);
  bf16_t* Wkb   = (bf16_t*)(w + 10485760);
  bf16_t* Wvb   = (bf16_t*)(w + 12582912);
  bf16_t* Wob   = (bf16_t*)(w + 14680064);
  bf16_t* fc1b  = (bf16_t*)(w + 16777216);
  bf16_t* fc2b  = (bf16_t*)(w + 25165824);
  bf16_t* qb    = (bf16_t*)(w + 33554432);
  bf16_t* kb    = (bf16_t*)(w + 41943040);
  bf16_t* vb    = (bf16_t*)(w + 50331648);
  bf16_t* ao    = (bf16_t*)(w + 58720256);
  float*  x1    = (float*) (w + 67108864);              // T*D f32
  bf16_t* h2    = (bf16_t*)(w + 83886080);
  bf16_t* mlp1  = (bf16_t*)(w + 92274688);              // T*MLP bf16

  float* out_x = (float*)d_out;
  float* out_p = (float*)d_out + T * Ddim;

  // weight casts
  cast_bf16_k<<<1024, 256, 0, stream>>>(Wq, Wqb, Ddim * Ddim);
  cast_bf16_k<<<1024, 256, 0, stream>>>(Wk, Wkb, Ddim * Ddim);
  cast_bf16_k<<<1024, 256, 0, stream>>>(Wv, Wvb, Ddim * Ddim);
  cast_bf16_k<<<1024, 256, 0, stream>>>(Wo, Wob, Ddim * Ddim);
  cast_bf16_k<<<4096, 256, 0, stream>>>(fc1_w, fc1b, Ddim * MLPdim);
  cast_bf16_k<<<4096, 256, 0, stream>>>(fc2_w, fc2b, MLPdim * Ddim);

  // LN1
  ln_bf16_k<<<(int)T, 256, 0, stream>>>(x, ln1_g, ln1_b, h1);

  // QKV projections  (out bf16 only)
  dim3 gq(Ddim / BN, (int)T / BM);
  gemm_bf16_k<0, 0, 0, 0, 1><<<gq, 256, 0, stream>>>(h1, Wqb, (int)T, Ddim, Ddim, nullptr, nullptr, nullptr, qb);
  gemm_bf16_k<0, 0, 0, 0, 1><<<gq, 256, 0, stream>>>(h1, Wkb, (int)T, Ddim, Ddim, nullptr, nullptr, nullptr, kb);
  gemm_bf16_k<0, 0, 0, 0, 1><<<gq, 256, 0, stream>>>(h1, Wvb, (int)T, Ddim, Ddim, nullptr, nullptr, nullptr, vb);

  // attention
  size_t smem = (size_t)16 * SROW * 4 + (size_t)8 * 64 * 40 * 2 + (size_t)16 * 64 * 4;
  attn_k<<<dim3(Sdim / 16, Bdim * Hdim), 256, smem, stream>>>(qb, kb, vb, out_p, ao);

  // output projection + residual(x) -> x1 (f32)
  gemm_bf16_k<0, 1, 0, 1, 0><<<gq, 256, 0, stream>>>(ao, Wob, (int)T, Ddim, Ddim, nullptr, x, x1, nullptr);

  // LN2
  ln_bf16_k<<<(int)T, 256, 0, stream>>>(x1, ln2_g, ln2_b, h2);

  // MLP: fc1 (+bias, gelu, bf16 out), fc2 (+bias, +residual, f32 out -> d_out)
  dim3 g1(MLPdim / BN, (int)T / BM);
  gemm_bf16_k<1, 0, 1, 0, 1><<<g1, 256, 0, stream>>>(h2, fc1b, (int)T, MLPdim, Ddim, fc1_b, nullptr, nullptr, mlp1);
  gemm_bf16_k<1, 1, 0, 1, 0><<<gq, 256, 0, stream>>>(mlp1, fc2b, (int)T, Ddim, MLPdim, fc2_b, x1, out_x, nullptr);
}
